// predictNE_14637248545193
// MI455X (gfx1250) — matrix operand order repaired and checked by reading, not compiled, not run on hardware
//
#include <hip/hip_runtime.h>

// LSTM: B=256, T=2048, I=12, H=64, 4H=256.
// 16 blocks x 128 threads (4 waves). Block owns 16 batch rows for all T.
// Wave hc owns hidden chunk [hc*16, hc*16+16) across all 4 gates.
// Per step: gates[16x256] = A[16x96] x B[96x256] via 12 v_wmma_f32_16x16x32_f16,
// A row = [h(64) | x_t(12) | 1 | 0...], B = [w_hh^T; w_ih^T; (b_ih+b_hh)^T; 0].
// FC head fused as 2 more WMMAs on wave 0. Ping-pong LDS panel -> 1 barrier/step.

typedef __attribute__((ext_vector_type(16))) _Float16 v16h;
typedef __attribute__((ext_vector_type(8)))  _Float16 v8h;
typedef __attribute__((ext_vector_type(8)))  float    v8f;

#define BATCH 256
#define TLEN  2048
#define IDIM  12
#define HDIM  64
#define KAUG  96   // 64 (h) + 12 (x) + 1 (bias) + 19 (pad)

// Unclamped fast forms: IEEE inf/rcp semantics give correct saturation
// (exp->inf => rcp->0 => sigmoid->{0,1}, tanh->±1).
__device__ __forceinline__ float fast_sigm(float v) {
    return __builtin_amdgcn_rcpf(1.f + __expf(-v));       // exp + add + rcp
}
__device__ __forceinline__ float fast_tanh(float v) {
    return 1.f - 2.f * __builtin_amdgcn_rcpf(__expf(2.f * v) + 1.f);
}

// A fragment per CDNA5 16-bit 16x32 A layout:
// lane<16:  halves 0..7 -> K=c*32+0..7,   halves 8..15 -> K=c*32+16..23 (m=lane)
// lane>=16: halves 0..7 -> K=c*32+8..15,  halves 8..15 -> K=c*32+24..31 (m=lane-16)
__device__ __forceinline__ v16h load_a_frag(const _Float16* p) {
    v8h lo = *(const v8h*)(p);
    v8h hi = *(const v8h*)(p + 16);
    v16h r;
#pragma unroll
    for (int j = 0; j < 8; ++j) { r[j] = lo[j]; r[j + 8] = hi[j]; }
    return r;
}

__global__ void __launch_bounds__(128)
lstm_wmma_kernel(const float* __restrict__ x,     // [B,T,I]
                 const float* __restrict__ w_ih,  // [4H,I]
                 const float* __restrict__ w_hh,  // [4H,H]
                 const float* __restrict__ b_ih,  // [4H]
                 const float* __restrict__ b_hh,  // [4H]
                 const float* __restrict__ fc_w,  // [1,H]
                 const float* __restrict__ fc_b,  // [1]
                 float* __restrict__ out)         // [B,T,1]
{
    __shared__ __align__(16) _Float16 Ap[2][16 * KAUG]; // ping-pong A panels

    const int tid   = threadIdx.x;
    const int lane  = tid & 31;
    const int hc    = tid >> 5;                 // wave id = hidden chunk 0..3
    const int bbase = blockIdx.x * 16;

    const int m_a   = lane & 15;                // A-row this lane feeds
    const int kb_a  = (lane & 16) ? 8 : 0;      // A K sub-offset
    const int kb_b  = (lane & 16) ? 8 : 0;      // B K sub-offset
    const int row_c = (lane & 16) ? 8 : 0;      // C/D M offset
    const int hcol  = hc * 16 + (lane & 15);    // hidden index this lane owns

    // ---- Preload gate B fragments (constant over T): 4 gates x 3 K-chunks ----
    // B[k][n] = Wcomb[n][k]; n = gi*64 + hc*16 + (lane&15)
    v16h bfrag[4][3];
#pragma unroll
    for (int gi = 0; gi < 4; ++gi) {
        const int n = gi * 64 + hc * 16 + (lane & 15);
#pragma unroll
        for (int c = 0; c < 3; ++c) {
            v16h f;
#pragma unroll
            for (int j = 0; j < 16; ++j) {
                const int k = c * 32 + kb_b + j + (j & 8);
                float v;
                if (k < HDIM)              v = w_hh[n * HDIM + k];
                else if (k < HDIM + IDIM)  v = w_ih[n * IDIM + (k - HDIM)];
                else if (k == HDIM + IDIM) v = b_ih[n] + b_hh[n];
                else                       v = 0.f;
                f[j] = (_Float16)v;
            }
            bfrag[gi][c] = f;
        }
    }

    // ---- FC head B fragments: fc_w in column n==0, zeros elsewhere ----
    v16h fb0, fb1;
    {
        const bool col0 = (lane & 15) == 0;
#pragma unroll
        for (int j = 0; j < 16; ++j) {
            const int k = kb_b + j + (j & 8);
            fb0[j] = (_Float16)(col0 ? fc_w[k] : 0.f);
            fb1[j] = (_Float16)(col0 ? fc_w[32 + k] : 0.f);
        }
    }

    // ---- Init both panels: h = 0, bias col = 1, pad = 0 ----
    for (int idx = tid; idx < 2 * 16 * KAUG; idx += 128) {
        const int col = idx % KAUG;
        ((_Float16*)Ap)[idx] = (col == HDIM + IDIM) ? (_Float16)1.0f
                                                    : (_Float16)0.0f;
    }
    __syncthreads();

    // ---- Fixed x-staging slots: element tid, and tid+128 (tid < 64) ----
    const int  m0 = tid / IDIM, i0 = tid % IDIM;
    const int  e1 = tid + 128;
    const int  m1 = e1 / IDIM, i1 = e1 % IDIM;
    const bool has1 = (e1 < 16 * IDIM);                      // tid < 64
    const float* xg0 = x + (size_t)(bbase + m0) * TLEN * IDIM + i0;
    const float* xg1 = x + (size_t)(bbase + m1) * TLEN * IDIM + i1;
    const int xoff0 = m0 * KAUG + HDIM + i0;
    const int xoff1 = m1 * KAUG + HDIM + i1;

    // stage x(t=0) into panel 0
    Ap[0][xoff0] = (_Float16)xg0[0];
    if (has1) Ap[0][xoff1] = (_Float16)xg1[0];
    xg0 += IDIM; xg1 += IDIM;
    __syncthreads();

    const float fcb = fc_b[0];
    const int afoff = m_a * KAUG + kb_a;        // this lane's A frag offset

    float cst[8];
#pragma unroll
    for (int r = 0; r < 8; ++r) cst[r] = 0.f;

    // prologue frags from panel 0: h(-1)=0, x(0)
    v16h a0 = load_a_frag(&Ap[0][afoff]);
    v16h a1 = load_a_frag(&Ap[0][afoff] + 32);
    v16h a2 = load_a_frag(&Ap[0][afoff] + 64);

    // One time step: gates from current frags; write h(t)/x(t+1) into panel w;
    // single barrier; reload frags from w (used by FC now, gates next step).
    auto step = [&](int t, _Float16* __restrict__ w) {
        v8f acc[4];
#pragma unroll
        for (int gi = 0; gi < 4; ++gi) {
            v8f z = {0.f, 0.f, 0.f, 0.f, 0.f, 0.f, 0.f, 0.f};
            z = __builtin_amdgcn_wmma_f32_16x16x32_f16(false, a0, false, bfrag[gi][0],
                                                       (short)0, z, false, false);
            z = __builtin_amdgcn_wmma_f32_16x16x32_f16(false, a1, false, bfrag[gi][1],
                                                       (short)0, z, false, false);
            z = __builtin_amdgcn_wmma_f32_16x16x32_f16(false, a2, false, bfrag[gi][2],
                                                       (short)0, z, false, false);
            acc[gi] = z;
        }

        float hnew[8];
#pragma unroll
        for (int r = 0; r < 8; ++r) {
            const float ig = fast_sigm(acc[0][r]);
            const float fg = fast_sigm(acc[1][r]);
            const float gg = fast_tanh(acc[2][r]);
            const float og = fast_sigm(acc[3][r]);
            cst[r] = fg * cst[r] + ig * gg;
            hnew[r] = og * fast_tanh(cst[r]);
        }

        // write h(t) into the *other* panel (read set untouched -> no barrier
        // needed before these writes)
#pragma unroll
        for (int r = 0; r < 8; ++r) {
            w[(r + row_c) * KAUG + hcol] = (_Float16)hnew[r];
        }
        if (t + 1 < TLEN) {
            w[xoff0] = (_Float16)xg0[0];
            if (has1) w[xoff1] = (_Float16)xg1[0];
            xg0 += IDIM; xg1 += IDIM;
        }
        __syncthreads();   // writers done; panel w = {h(t), x(t+1)}

        a0 = load_a_frag(w + afoff);
        a1 = load_a_frag(w + afoff + 32);
        a2 = load_a_frag(w + afoff + 64);

        // FC head on wave 0: out[b,t] = h(t).fc_w + fc_b via 2 WMMAs
        if (hc == 0) {                       // wave-uniform: EXEC all-1s
            v8f z = {0.f, 0.f, 0.f, 0.f, 0.f, 0.f, 0.f, 0.f};
            z = __builtin_amdgcn_wmma_f32_16x16x32_f16(false, a0, false, fb0,
                                                       (short)0, z, false, false);
            z = __builtin_amdgcn_wmma_f32_16x16x32_f16(false, a1, false, fb1,
                                                       (short)0, z, false, false);
            if ((lane & 15) == 0) {          // lanes 0,16 hold D column N=0
                float* op = out + (size_t)(bbase + row_c) * TLEN + t;
#pragma unroll
                for (int r = 0; r < 8; ++r)
                    op[(size_t)r * TLEN] = z[r] + fcb;
            }
        }
        // panel w is next written at t+2, i.e. after every thread passes the
        // barrier inside step(t+1): the frag reloads above are race-free.
    };

    for (int t = 0; t < TLEN; t += 2) {
        step(t,     &Ap[1][0]);
        step(t + 1, &Ap[0][0]);
    }
}

extern "C" void kernel_launch(void* const* d_in, const int* in_sizes, int n_in,
                              void* d_out, int out_size, void* d_ws, size_t ws_size,
                              hipStream_t stream) {
    const float* x    = (const float*)d_in[0];
    const float* w_ih = (const float*)d_in[1];
    const float* w_hh = (const float*)d_in[2];
    const float* b_ih = (const float*)d_in[3];
    const float* b_hh = (const float*)d_in[4];
    const float* fc_w = (const float*)d_in[5];
    const float* fc_b = (const float*)d_in[6];
    float* out = (float*)d_out;

    lstm_wmma_kernel<<<dim3(BATCH / 16), dim3(128), 0, stream>>>(
        x, w_ih, w_hh, b_ih, b_hh, fc_w, fc_b, out);
}
